// QEffGrok1MultiHeadAttention_52931176955913
// MI455X (gfx1250) — matrix operand order repaired and checked by reading, not compiled, not run on hardware
//
#include <hip/hip_runtime.h>
#include <hip/hip_bf16.h>

typedef __attribute__((ext_vector_type(16))) _Float16 v16h;
typedef __attribute__((ext_vector_type(4)))  _Float16 v4h;
typedef __attribute__((ext_vector_type(8)))  float    v8f;
typedef __attribute__((ext_vector_type(4)))  float    v4f;
typedef __attribute__((ext_vector_type(4)))  unsigned int v4u;
typedef __attribute__((ext_vector_type(4)))  int      v4i;
typedef __attribute__((ext_vector_type(8)))  int      v8i;

#define B_SZ   2
#define S_LEN  2048
#define D_MOD  4096
#define NH     32
#define NKVH   8
#define HD     128
#define ATTN_MULT 0.08838834764831845f
#define MAX_ATTN  30.0f
#define MIN_MASKED -65504.0f

#if __has_builtin(__builtin_amdgcn_tensor_load_to_lds) && \
    __has_builtin(__builtin_amdgcn_s_wait_tensorcnt)
#define HAVE_TDM 1
#endif

// ---------------------------------------------------------------------------
// WMMA fragment helpers (wave32, v_wmma_f32_16x16x32_f16)
// A 16x32 (MxK) row-major src: a[2g],a[2g+1]=A[m][k], k=2*(g&3)+16*(g>>2)+8*hi
// B 32x16 (KxN) from [n][k]-major src: b[e] = B[k=(e&7)+8*hi+16*(e>>3)][n=lo]
// C/D 16x16 f32: elem r -> row (r + 8*hi), col lo
// ---------------------------------------------------------------------------
__device__ __forceinline__ v16h load_a_frag(const _Float16* __restrict__ p, int ld) {
  const int lane = threadIdx.x & 31, lo = lane & 15, hi = lane >> 4;
  v16h a;
#pragma unroll
  for (int g = 0; g < 8; ++g) {
    const int k = 2 * (g & 3) + 16 * (g >> 2) + 8 * hi;
    a[2 * g]     = p[lo * ld + k];
    a[2 * g + 1] = p[lo * ld + k + 1];
  }
  return a;
}

__device__ __forceinline__ v16h load_b_frag_t(const _Float16* __restrict__ p, int ld) {
  const int lane = threadIdx.x & 31, lo = lane & 15, hi = lane >> 4;
  v16h b;
#pragma unroll
  for (int e = 0; e < 16; ++e) b[e] = p[lo * ld + (e & 7) + 8 * hi + 16 * (e >> 3)];
  return b;
}

__device__ __forceinline__ v8f wmma_f16(v16h a, v16h b, v8f c) {
  return __builtin_amdgcn_wmma_f32_16x16x32_f16(false, a, false, b, (short)0, c,
                                                false, false);
}

__device__ __forceinline__ v4h load4_cvt(const float* __restrict__ p) {
  const v4f t = *(const v4f*)p;
  v4h r;
  r[0] = (_Float16)t[0]; r[1] = (_Float16)t[1];
  r[2] = (_Float16)t[2]; r[3] = (_Float16)t[3];
  return r;
}
__device__ __forceinline__ v4h load4_cvt(const _Float16* __restrict__ p) {
  return *(const v4h*)p;
}

// ---------------------------------------------------------------------------
// TDM: DMA a 2-D 16-bit tile (128 elems per row, `rows` rows) into LDS with
// +4-half row padding (pad_interval=256B, pad_amount=2 DWORDs -> LDS ld=132).
// ---------------------------------------------------------------------------
#ifdef HAVE_TDM
__device__ __forceinline__ void tdm_load_tile_2d(unsigned lds_off,
                                                 const void* gaddr, int rows,
                                                 int row_stride_elems) {
  const unsigned long long ga = (unsigned long long)(size_t)gaddr;
  v4u g0;
  g0[0] = 1u;                                   // count = 1 (valid descriptor)
  g0[1] = lds_off;                              // lds_addr (bytes)
  g0[2] = (unsigned)(ga & 0xFFFFFFFFu);         // global_addr[31:0]
  g0[3] = (unsigned)((ga >> 32) & 0x1FFFFFFu)   // global_addr[56:32]
        | (2u << 30);                           // type = 2 ("image")
  v8i g1;
  g1[0] = (1 << 16)      // data_size = 2 bytes
        | (1 << 20)      // pad_enable
        | (5 << 22)      // pad_interval: 32x8B = 256B (one 128-half row)
        | (1 << 25);     // pad_amount: 2 DWORDs (= 4 halves)
  g1[1] = (128 & 0xFFFF) << 16;                 // tensor_dim0 lo16 (=128)
  g1[2] = (rows & 0xFFFF) << 16;                // dim0 hi16=0 | tensor_dim1 lo16
  g1[3] = (128 & 0xFFFF) << 16;                 // dim1 hi16=0 | tile_dim0 (=128)
  g1[4] = rows & 0xFFFF;                        // tile_dim1 | tile_dim2=0
  g1[5] = row_stride_elems;                     // tensor_dim0_stride lo32
  g1[6] = 0;
  g1[7] = 0;
  const v4i z = {0, 0, 0, 0};
#if __clang_major__ >= 23
  const v8i z8 = {0, 0, 0, 0, 0, 0, 0, 0};
  __builtin_amdgcn_tensor_load_to_lds(g0, g1, z, z, z8, 0);
#else
  __builtin_amdgcn_tensor_load_to_lds(g0, g1, z, z, 0);
#endif
}
#endif

// ---------------------------------------------------------------------------
// Tiled GEMM: C[M,N] = A[M,K] * W[K,N]; block tile 128x128, k-step 32.
// 8 waves (4M x 2N), each wave 32x64 -> 8 WMMAs per k-step.
// ---------------------------------------------------------------------------
template <typename TIN, typename TOUT>
__global__ __launch_bounds__(256) void gemm_kernel(const TIN* __restrict__ A,
                                                   const float* __restrict__ W,
                                                   TOUT* __restrict__ C, int M,
                                                   int N, int K) {
  __shared__ _Float16 As[128][40];
  __shared__ _Float16 Bt[128][40];  // [n][k]

  const int tid = threadIdx.x;
  const int wave = tid >> 5;
  const int wm = wave & 3, wn = wave >> 2;
  const int m0 = blockIdx.y * 128, n0 = blockIdx.x * 128;
  const int lane = tid & 31, lo = lane & 15, hi = lane >> 4;

  v8f acc[2][4] = {};

  for (int k0 = 0; k0 < K; k0 += 32) {
    if (k0 + 32 < K)  // global_prefetch_b8
      __builtin_prefetch(&A[(size_t)(m0 + (tid >> 1)) * K + k0 + 32 + (tid & 1) * 16], 0, 0);

    // A tile: 128 rows x 32 k (vectorized x4)
#pragma unroll
    for (int i = tid; i < 1024; i += 256) {
      const int r = i >> 3, c4 = (i & 7) * 4;
      *(v4h*)&As[r][c4] = load4_cvt(&A[(size_t)(m0 + r) * K + k0 + c4]);
    }
    // W tile -> Bt[n][k]: 32 k x 128 n (vectorized x4 over n)
#pragma unroll
    for (int i = tid; i < 1024; i += 256) {
      const int k = i >> 5, n4 = (i & 31) * 4;
      const v4f w = *(const v4f*)&W[(size_t)(k0 + k) * N + n0 + n4];
      Bt[n4 + 0][k] = (_Float16)w[0];
      Bt[n4 + 1][k] = (_Float16)w[1];
      Bt[n4 + 2][k] = (_Float16)w[2];
      Bt[n4 + 3][k] = (_Float16)w[3];
    }
    __syncthreads();

#pragma unroll
    for (int mi = 0; mi < 2; ++mi) {
      const v16h a = load_a_frag(&As[wm * 32 + mi * 16][0], 40);
#pragma unroll
      for (int ni = 0; ni < 4; ++ni) {
        const v16h b = load_b_frag_t(&Bt[wn * 64 + ni * 16][0], 40);
        acc[mi][ni] = wmma_f16(a, b, acc[mi][ni]);
      }
    }
    __syncthreads();
  }

#pragma unroll
  for (int mi = 0; mi < 2; ++mi)
#pragma unroll
    for (int ni = 0; ni < 4; ++ni)
#pragma unroll
      for (int r = 0; r < 8; ++r) {
        const int row = m0 + wm * 32 + mi * 16 + r + 8 * hi;
        const int col = n0 + wn * 64 + ni * 16 + lo;
        C[(size_t)row * N + col] = (TOUT)acc[mi][ni][r];
      }
}

// ---------------------------------------------------------------------------
// RoPE in place on f16 [B*S, nheads*128]
// ---------------------------------------------------------------------------
__global__ void rope_kernel(_Float16* __restrict__ X,
                            const long long* __restrict__ pos_ids, int nheads) {
  const size_t idx = (size_t)blockIdx.x * blockDim.x + threadIdx.x;
  const size_t total = (size_t)B_SZ * S_LEN * nheads * 64;
  if (idx >= total) return;
  const int i = (int)(idx & 63);
  const size_t th = idx >> 6;
  const int head = (int)(th % nheads);
  const size_t tok = th / nheads;
  const float pos = (float)pos_ids[tok];
  const float f = pos * __powf(10000.0f, -(float)i * (1.0f / 64.0f));
  const float c = __cosf(f), s = __sinf(f);
  const size_t base = tok * ((size_t)nheads * HD) + (size_t)head * HD + i;
  const float x1 = (float)X[base], x2 = (float)X[base + 64];
  X[base]      = (_Float16)(x1 * c - x2 * s);
  X[base + 64] = (_Float16)(x2 * c + x1 * s);
}

// ---------------------------------------------------------------------------
// Flash attention, causal, tanh soft-cap. grid = (B*NH, S/128), block = 256.
// K/V tiles (32 keys x 128 d, f16) staged by the Tensor Data Mover; V is then
// transposed LDS->LDS so PV B-fragments are contiguous wide DS reads.
// ---------------------------------------------------------------------------
#define LDKK 132
#define LDVV 36
#define LDPP 36

__global__ __launch_bounds__(256) void flash_attn_kernel(
    const _Float16* __restrict__ Q, const _Float16* __restrict__ Kmat,
    const _Float16* __restrict__ V, _Float16* __restrict__ Out) {
  __shared__ _Float16 Klds[32][LDKK];      // [key][d]
#ifdef HAVE_TDM
  __shared__ _Float16 Vlds[32][LDKK];      // [key][d] (TDM landing buffer)
#endif
  __shared__ _Float16 Vt[128][LDVV];       // [d][key]
  __shared__ _Float16 Plds[8][16][LDPP];   // per-wave P staging

  const int bh = blockIdx.x;
  const int b = bh >> 5, h = bh & 31, kvh = h >> 2;  // GQA: 4 q-heads / kv-head
  const int qbase = blockIdx.y * 128;
  const int tid = threadIdx.x, wave = tid >> 5, lane = tid & 31;
  const int lo = lane & 15, hi = lane >> 4;
  const int qrow0 = qbase + wave * 16;

  const _Float16* qptr =
      Q + ((size_t)(b * S_LEN + qrow0)) * (NH * HD) + (size_t)h * HD;
  v16h qa[4];
#pragma unroll
  for (int j = 0; j < 4; ++j) qa[j] = load_a_frag(qptr + j * 32, NH * HD);

  v8f O[8] = {};
  float mrow[8], lrow[8];
#pragma unroll
  for (int r = 0; r < 8; ++r) { mrow[r] = -1e30f; lrow[r] = 0.0f; }

#ifdef HAVE_TDM
  const unsigned klds_off = (unsigned)(size_t)(void*)&Klds[0][0];
  const unsigned vlds_off = (unsigned)(size_t)(void*)&Vlds[0][0];
#endif

  const int nkt = (qbase + 128) >> 5;
  for (int kt = 0; kt < nkt; ++kt) {
    const int kbase = kt << 5;
#ifdef HAVE_TDM
    if (wave == 0) {
      const size_t g0 =
          ((size_t)(b * S_LEN + kbase)) * (NKVH * HD) + (size_t)kvh * HD;
      tdm_load_tile_2d(klds_off, Kmat + g0, 32, NKVH * HD);
      tdm_load_tile_2d(vlds_off, V + g0, 32, NKVH * HD);
      __builtin_amdgcn_s_wait_tensorcnt(0);
    }
    __syncthreads();
    // LDS->LDS transpose of V: each thread moves a 1x16 sliver
    {
      const int key = tid >> 3;
      const int d0 = (tid & 7) * 16;
#pragma unroll
      for (int j = 0; j < 16; ++j) Vt[d0 + j][key] = Vlds[key][d0 + j];
    }
#else
    for (int i = tid; i < 32 * 128; i += 256) {
      const int key = i >> 7, d = i & 127;
      const size_t g =
          ((size_t)(b * S_LEN + kbase + key)) * (NKVH * HD) + (size_t)kvh * HD + d;
      Klds[key][d] = Kmat[g];
      Vt[d][key]   = V[g];
    }
#endif
    __syncthreads();

    if (kbase <= qrow0 + 15) {
      v8f S0 = {}, S1 = {};
#pragma unroll
      for (int j = 0; j < 4; ++j) {
        const v16h b0 = load_b_frag_t(&Klds[0][j * 32], LDKK);
        const v16h b1 = load_b_frag_t(&Klds[16][j * 32], LDKK);
        S0 = wmma_f16(qa[j], b0, S0);
        S1 = wmma_f16(qa[j], b1, S1);
      }
#pragma unroll
      for (int r = 0; r < 8; ++r) {
        const int qrow = qrow0 + r + 8 * hi;
        float s0 = S0[r] * ATTN_MULT;
        s0 = MAX_ATTN * tanhf(s0 * (1.0f / MAX_ATTN));
        if (kbase + lo > qrow) s0 = MIN_MASKED;
        float s1 = S1[r] * ATTN_MULT;
        s1 = MAX_ATTN * tanhf(s1 * (1.0f / MAX_ATTN));
        if (kbase + 16 + lo > qrow) s1 = MIN_MASKED;

        float rm = fmaxf(s0, s1);
#pragma unroll
        for (int o = 1; o < 16; o <<= 1) rm = fmaxf(rm, __shfl_xor(rm, o, 32));
        const float mnew = fmaxf(mrow[r], rm);
        const float corr = __expf(mrow[r] - mnew);
        const float p0 = __expf(s0 - mnew);
        const float p1 = __expf(s1 - mnew);
        float rs = p0 + p1;
#pragma unroll
        for (int o = 1; o < 16; o <<= 1) rs += __shfl_xor(rs, o, 32);
        lrow[r] = lrow[r] * corr + rs;
        mrow[r] = mnew;
#pragma unroll
        for (int j = 0; j < 8; ++j) O[j][r] = O[j][r] * corr;
        Plds[wave][r + 8 * hi][lo]      = (_Float16)p0;
        Plds[wave][r + 8 * hi][16 + lo] = (_Float16)p1;
      }
      // P (C-layout) -> A-layout via per-wave LDS (DS ops in-order per wave)
      const v16h pa = load_a_frag(&Plds[wave][0][0], LDPP);
#pragma unroll
      for (int j = 0; j < 8; ++j) {
        const v16h vb = load_b_frag_t(&Vt[j * 16][0], LDVV);
        O[j] = wmma_f16(pa, vb, O[j]);
      }
    }
    __syncthreads();
  }

#pragma unroll
  for (int r = 0; r < 8; ++r) {
    const int qrow = qrow0 + r + 8 * hi;
    const float inv = 1.0f / lrow[r];
    const size_t base =
        ((size_t)(b * S_LEN + qrow)) * (NH * HD) + (size_t)h * HD;
#pragma unroll
    for (int j = 0; j < 8; ++j)
      Out[base + j * 16 + lo] = (_Float16)(O[j][r] * inv);
  }
}

// ---------------------------------------------------------------------------
extern "C" void kernel_launch(void* const* d_in, const int* in_sizes, int n_in,
                              void* d_out, int out_size, void* d_ws,
                              size_t ws_size, hipStream_t stream) {
  (void)in_sizes; (void)n_in; (void)out_size; (void)ws_size;
  const float* hs = (const float*)d_in[0];
  // d_in[1] = attention_mask (bool, causal) — applied analytically
  const long long* pos = (const long long*)d_in[2];
  const float* Wq = (const float*)d_in[3];
  const float* Wk = (const float*)d_in[4];
  const float* Wv = (const float*)d_in[5];
  const float* Wo = (const float*)d_in[6];
  float* out = (float*)d_out;

  const size_t M = (size_t)B_SZ * S_LEN;  // 4096
  _Float16* Qws = (_Float16*)d_ws;                 // M * 4096
  _Float16* Kws = Qws + M * (NH * HD);             // M * 1024
  _Float16* Vws = Kws + M * (NKVH * HD);           // M * 1024
  _Float16* Aws = Vws + M * (NKVH * HD);           // M * 4096

  dim3 blk(256);
  // QKV projections
  gemm_kernel<float, _Float16><<<dim3(4096 / 128, M / 128), blk, 0, stream>>>(
      hs, Wq, Qws, (int)M, NH * HD, D_MOD);
  gemm_kernel<float, _Float16><<<dim3(1024 / 128, M / 128), blk, 0, stream>>>(
      hs, Wk, Kws, (int)M, NKVH * HD, D_MOD);
  gemm_kernel<float, _Float16><<<dim3(1024 / 128, M / 128), blk, 0, stream>>>(
      hs, Wv, Vws, (int)M, NKVH * HD, D_MOD);

  // RoPE on Q and K
  {
    const size_t tq = M * NH * 64;
    rope_kernel<<<dim3((unsigned)((tq + 255) / 256)), blk, 0, stream>>>(Qws, pos, NH);
    const size_t tk = M * NKVH * 64;
    rope_kernel<<<dim3((unsigned)((tk + 255) / 256)), blk, 0, stream>>>(Kws, pos, NKVH);
  }

  // Flash attention
  flash_attn_kernel<<<dim3(B_SZ * NH, S_LEN / 128), blk, 0, stream>>>(Qws, Kws,
                                                                      Vws, Aws);

  // Output projection -> f32
  gemm_kernel<_Float16, float><<<dim3(4096 / 128, M / 128), blk, 0, stream>>>(
      Aws, Wo, out, (int)M, D_MOD, NH * HD);
}
